// MSARowAttentionWithPairBias_26783416058533
// MI455X (gfx1250) — matrix operand order repaired and checked by reading, not compiled, not run on hardware
//
#include <hip/hip_runtime.h>
#include <hip/hip_bf16.h>

typedef __attribute__((ext_vector_type(16))) _Float16 v16h;
typedef __attribute__((ext_vector_type(8)))  float    v8f;
typedef __attribute__((ext_vector_type(8)))  _Float16 h8;

#define WMMA_F16(A, B, C) \
  __builtin_amdgcn_wmma_f32_16x16x32_f16(false, (A), false, (B), (short)0, (C), false, false)

// Problem constants
#define NROW 128          // MSA rows N
#define LSEQ 256          // sequence length L
#define CM   256          // c_m
#define CZ   128          // c_z
#define NH   8            // heads
#define HD   32           // head dim
#define EQKV 768          // 3*CM
#define MTOT 32768        // N*L rows
#define QSCALE 0.17677669529663687f  // 1/sqrt(32)

// ---------------------------------------------------------------------------
// Kernel: convert f32 weight (E,C) row-major -> f16 transposed (C,E) row-major
// ---------------------------------------------------------------------------
__global__ void convert_wT_kernel(const float* __restrict__ src, _Float16* __restrict__ dst,
                                  int E, int C) {
  int idx = blockIdx.x * blockDim.x + threadIdx.x;
  if (idx >= E * C) return;
  int e = idx / C;
  int c = idx - e * C;
  dst[c * E + e] = (_Float16)src[idx];
}

// ---------------------------------------------------------------------------
// Kernel: pair bias  bias[h][i][j] = sum_c LN(z[i,j,:])[c] * w_pair[h][c]
// block = 128 threads, one (i,j) per thread
// ---------------------------------------------------------------------------
__global__ __launch_bounds__(128) void pair_bias_kernel(
    const float* __restrict__ z, const float* __restrict__ g, const float* __restrict__ b,
    const float* __restrict__ wp, float* __restrict__ bias) {
  __shared__ float swp[NH * CZ];
  __shared__ float sg[CZ];
  __shared__ float sb[CZ];
  int tid = threadIdx.x;
  for (int i = tid; i < NH * CZ; i += 128) swp[i] = wp[i];
  sg[tid] = g[tid];
  sb[tid] = b[tid];
  __syncthreads();

  int idx = blockIdx.x * 128 + tid;        // flat (i,j), i = idx>>8, j = idx&255
  const float* zr = z + (size_t)idx * CZ;

  float sum = 0.f, ss = 0.f;
#pragma unroll
  for (int c = 0; c < CZ; c += 4) {
    float4 v = *(const float4*)(zr + c);
    sum += v.x + v.y + v.z + v.w;
    ss  += v.x * v.x + v.y * v.y + v.z * v.z + v.w * v.w;
  }
  float mu   = sum * (1.f / CZ);
  float var  = ss * (1.f / CZ) - mu * mu;
  float rinv = rsqrtf(var + 1e-5f);

  float acc[NH];
#pragma unroll
  for (int h = 0; h < NH; ++h) acc[h] = 0.f;

#pragma unroll 4
  for (int c = 0; c < CZ; c += 4) {
    float4 v = *(const float4*)(zr + c);
    float zn0 = (v.x - mu) * rinv * sg[c + 0] + sb[c + 0];
    float zn1 = (v.y - mu) * rinv * sg[c + 1] + sb[c + 1];
    float zn2 = (v.z - mu) * rinv * sg[c + 2] + sb[c + 2];
    float zn3 = (v.w - mu) * rinv * sg[c + 3] + sb[c + 3];
#pragma unroll
    for (int h = 0; h < NH; ++h) {
      const float* w = &swp[h * CZ + c];
      acc[h] += zn0 * w[0] + zn1 * w[1] + zn2 * w[2] + zn3 * w[3];
    }
  }
  int i = idx >> 8, j = idx & 255;
#pragma unroll
  for (int h = 0; h < NH; ++h)
    bias[((size_t)h * LSEQ + i) * LSEQ + j] = acc[h];
}

// ---------------------------------------------------------------------------
// Kernel: layernorm(m) -> f16, one wave32 per 256-channel row
// ---------------------------------------------------------------------------
__global__ __launch_bounds__(256) void ln_m_kernel(
    const float* __restrict__ m, const float* __restrict__ g, const float* __restrict__ b,
    _Float16* __restrict__ mn) {
  int tid = threadIdx.x, lane = tid & 31, w = tid >> 5;
  int row = blockIdx.x * 8 + w;            // MTOT rows
  const float* src = m + (size_t)row * CM + lane * 8;
  float4 a0 = *(const float4*)src;
  float4 a1 = *(const float4*)(src + 4);
  float sum = a0.x + a0.y + a0.z + a0.w + a1.x + a1.y + a1.z + a1.w;
  float ss  = a0.x * a0.x + a0.y * a0.y + a0.z * a0.z + a0.w * a0.w +
              a1.x * a1.x + a1.y * a1.y + a1.z * a1.z + a1.w * a1.w;
#pragma unroll
  for (int off = 1; off < 32; off <<= 1) {
    sum += __shfl_xor(sum, off, 32);
    ss  += __shfl_xor(ss, off, 32);
  }
  float mu   = sum * (1.f / CM);
  float var  = ss * (1.f / CM) - mu * mu;
  float rinv = rsqrtf(var + 1e-5f);
  float4 g0 = *(const float4*)(g + lane * 8);
  float4 g1 = *(const float4*)(g + lane * 8 + 4);
  float4 b0 = *(const float4*)(b + lane * 8);
  float4 b1 = *(const float4*)(b + lane * 8 + 4);
  h8 o;
  o[0] = (_Float16)((a0.x - mu) * rinv * g0.x + b0.x);
  o[1] = (_Float16)((a0.y - mu) * rinv * g0.y + b0.y);
  o[2] = (_Float16)((a0.z - mu) * rinv * g0.z + b0.z);
  o[3] = (_Float16)((a0.w - mu) * rinv * g0.w + b0.w);
  o[4] = (_Float16)((a1.x - mu) * rinv * g1.x + b1.x);
  o[5] = (_Float16)((a1.y - mu) * rinv * g1.y + b1.y);
  o[6] = (_Float16)((a1.z - mu) * rinv * g1.z + b1.z);
  o[7] = (_Float16)((a1.w - mu) * rinv * g1.w + b1.w);
  *(h8*)(mn + (size_t)row * CM + lane * 8) = o;
}

// ---------------------------------------------------------------------------
// Kernel: QKV GEMM  qkv[m][e] = mn[m][:] . Wt[:, e] + b[e], q cols scaled
// block tile 128(M) x 64(N), 8 waves each computing 32x32, K-step 32
// ---------------------------------------------------------------------------
__global__ __launch_bounds__(256) void qkv_gemm_kernel(
    const _Float16* __restrict__ A,   // [MTOT][CM]
    const _Float16* __restrict__ Bt,  // [CM][EQKV] (K rows)
    const float* __restrict__ bvec,   // [EQKV]
    _Float16* __restrict__ Cmat) {    // [MTOT][EQKV]
  __shared__ __attribute__((aligned(16))) _Float16 Alds[128 * 40];
  __shared__ __attribute__((aligned(16))) _Float16 Blds[64 * 40];  // transposed [n][k]
  int tid = threadIdx.x, lane = tid & 31, w = tid >> 5;
  int wm = w >> 1, wn = w & 1;
  int m0 = blockIdx.y * 128, n0 = blockIdx.x * 64;

  v8f acc[2][2];
#pragma unroll
  for (int i = 0; i < 2; ++i)
#pragma unroll
    for (int j = 0; j < 2; ++j)
#pragma unroll
      for (int r = 0; r < 8; ++r) acc[i][j][r] = 0.f;

  int ar = tid >> 1, ah = (tid & 1) * 16;
  int br = tid >> 3, bc = (tid & 7) * 8;
  int rl = lane & 15, ksel8 = (lane >> 4) << 3, ksel16 = (lane >> 4) << 4;

  for (int k0 = 0; k0 < CM; k0 += 32) {
    __syncthreads();
    {  // stage A tile 128x32
      const _Float16* s = A + (size_t)(m0 + ar) * CM + k0 + ah;
      uint4 t0 = *(const uint4*)s;
      uint4 t1 = *(const uint4*)(s + 8);
      *(uint4*)&Alds[ar * 40 + ah]     = t0;
      *(uint4*)&Alds[ar * 40 + ah + 8] = t1;
    }
    {  // stage B tile 32x64 transposed -> [n][k]
      union { uint4 u; _Float16 h[8]; } t;
      t.u = *(const uint4*)(Bt + (size_t)(k0 + br) * EQKV + n0 + bc);
#pragma unroll
      for (int u = 0; u < 8; ++u) Blds[(bc + u) * 40 + br] = t.h[u];
    }
    __syncthreads();

    v16h af[2], bf[2];
#pragma unroll
    for (int i = 0; i < 2; ++i) {
      const _Float16* p = &Alds[(wm * 32 + i * 16 + rl) * 40];
#pragma unroll
      for (int jj = 0; jj < 8; ++jj) { af[i][jj] = p[ksel8 + jj]; af[i][8 + jj] = p[16 + ksel8 + jj]; }
    }
#pragma unroll
    for (int j = 0; j < 2; ++j) {
      const _Float16* p = &Blds[(wn * 32 + j * 16 + rl) * 40];
#pragma unroll
      for (int t = 0; t < 8; ++t) { bf[j][t] = p[ksel8 + t]; bf[j][8 + t] = p[16 + ksel8 + t]; }
    }
#pragma unroll
    for (int i = 0; i < 2; ++i)
#pragma unroll
      for (int j = 0; j < 2; ++j)
        acc[i][j] = WMMA_F16(af[i], bf[j], acc[i][j]);
  }

  int hi8 = (lane >> 4) * 8;
#pragma unroll
  for (int i = 0; i < 2; ++i)
#pragma unroll
    for (int j = 0; j < 2; ++j) {
      int colg = n0 + wn * 32 + j * 16 + rl;
      float bv  = bvec[colg];
      float scl = (colg < CM) ? QSCALE : 1.0f;   // fold 1/sqrt(D) into q
#pragma unroll
      for (int rr = 0; rr < 8; ++rr) {
        int rowg = m0 + wm * 32 + i * 16 + rr + hi8;
        Cmat[(size_t)rowg * EQKV + colg] = (_Float16)((acc[i][j][rr] + bv) * scl);
      }
    }
}

// ---------------------------------------------------------------------------
// Kernel: flash attention per (n,h). 8 waves, 2 q-tiles (16 rows) per wave,
// j-loop in steps of 32. D=32 == WMMA K-dim.
// ---------------------------------------------------------------------------
__global__ __launch_bounds__(256) void attn_kernel(
    const _Float16* __restrict__ qkv,  // [MTOT][EQKV]
    const float* __restrict__ bias,    // [NH][LSEQ][LSEQ]
    _Float16* __restrict__ o) {        // [MTOT][CM]
  __shared__ __attribute__((aligned(16))) _Float16 Klds[32 * 40];      // [j][d]
  __shared__ __attribute__((aligned(16))) _Float16 Vt[32 * 40];        // [d][j]
  __shared__ __attribute__((aligned(16))) _Float16 Plds[8 * 16 * 40];  // per-wave [row][j]

  int nh = blockIdx.x;
  int n = nh >> 3, h = nh & 7;
  int tid = threadIdx.x, lane = tid & 31, w = tid >> 5;
  int cl = lane & 15, ksel8 = (lane >> 4) << 3, ksel16 = (lane >> 4) << 4, hi8 = (lane >> 4) * 8;

  const _Float16* qbase = qkv + (size_t)n * LSEQ * EQKV + h * HD;
  const _Float16* kbase = qbase + CM;
  const _Float16* vbase = qbase + 2 * CM;
  const float* bbase = bias + (size_t)h * LSEQ * LSEQ;

  // load the two Q fragments (rows fixed for whole kernel)
  v16h qf[2];
#pragma unroll
  for (int qt = 0; qt < 2; ++qt) {
    int qrow0 = (w + qt * 8) * 16;
    const _Float16* p = qbase + (size_t)(qrow0 + cl) * EQKV;
#pragma unroll
    for (int j = 0; j < 8; ++j) { qf[qt][j] = p[ksel8 + j]; qf[qt][8 + j] = p[16 + ksel8 + j]; }
  }

  float mrow[2][8], lrow[2][8];
  v8f oa[2], ob[2];
#pragma unroll
  for (int qt = 0; qt < 2; ++qt)
#pragma unroll
    for (int r = 0; r < 8; ++r) { mrow[qt][r] = -1e30f; lrow[qt][r] = 0.f; oa[qt][r] = 0.f; ob[qt][r] = 0.f; }

  int jr = tid >> 3, ch = (tid & 7) * 4;  // staging coords

  for (int j0 = 0; j0 < LSEQ; j0 += 32) {
    __syncthreads();
    {  // stage K [j][d]
      *(uint2*)&Klds[jr * 40 + ch] = *(const uint2*)(kbase + (size_t)(j0 + jr) * EQKV + ch);
      // stage V transposed [d][j]
      union { uint2 u; _Float16 hh[4]; } t;
      t.u = *(const uint2*)(vbase + (size_t)(j0 + jr) * EQKV + ch);
#pragma unroll
      for (int u = 0; u < 4; ++u) Vt[(ch + u) * 40 + jr] = t.hh[u];
    }
    __syncthreads();

#pragma unroll
    for (int qt = 0; qt < 2; ++qt) {
      int qrow0 = (w + qt * 8) * 16;
      // S = Q Kt : two 16x16 tiles (cols j0..15, j0+16..31)
      v16h b0, b1;
      {
        const _Float16* p0 = &Klds[cl * 40];
        const _Float16* p1 = &Klds[(16 + cl) * 40];
#pragma unroll
        for (int t = 0; t < 8; ++t) { b0[t] = p0[ksel8 + t]; b0[8 + t] = p0[16 + ksel8 + t]; b1[t] = p1[ksel8 + t]; b1[8 + t] = p1[16 + ksel8 + t]; }
      }
      v8f s0, s1;
#pragma unroll
      for (int r = 0; r < 8; ++r) { s0[r] = 0.f; s1[r] = 0.f; }
      s0 = WMMA_F16(qf[qt], b0, s0);
      s1 = WMMA_F16(qf[qt], b1, s1);

      // add pair bias
      const float* bp = bbase + (size_t)(qrow0 + hi8) * LSEQ + j0 + cl;
#pragma unroll
      for (int r = 0; r < 8; ++r) {
        s0[r] += bp[r * LSEQ];
        s1[r] += bp[r * LSEQ + 16];
      }

      // online softmax (rows striped over 16-lane halves)
#pragma unroll
      for (int r = 0; r < 8; ++r) {
        float t = fmaxf(s0[r], s1[r]);
#pragma unroll
        for (int off = 1; off < 16; off <<= 1) t = fmaxf(t, __shfl_xor(t, off, 32));
        float mo = mrow[qt][r];
        float mn2 = fmaxf(mo, t);
        float sc = __expf(mo - mn2);
        mrow[qt][r] = mn2;
        lrow[qt][r] *= sc;
        oa[qt][r] *= sc;
        ob[qt][r] *= sc;
        float p0 = __expf(s0[r] - mn2);
        float p1 = __expf(s1[r] - mn2);
        s0[r] = p0; s1[r] = p1;
        float rs = p0 + p1;
#pragma unroll
        for (int off = 1; off < 16; off <<= 1) rs += __shfl_xor(rs, off, 32);
        lrow[qt][r] += rs;
      }

      // write P to this wave's LDS region, reload as A fragment
      _Float16* pp = &Plds[w * 16 * 40];
#pragma unroll
      for (int r = 0; r < 8; ++r) {
        int row = r + hi8;
        pp[row * 40 + cl]      = (_Float16)s0[r];
        pp[row * 40 + 16 + cl] = (_Float16)s1[r];
      }
      v16h pf;
      {
        const _Float16* p = pp + cl * 40;
#pragma unroll
        for (int j = 0; j < 8; ++j) { pf[j] = p[ksel8 + j]; pf[8 + j] = p[16 + ksel8 + j]; }
      }
      // O += P V : B fragments from Vt rows (contiguous)
      v16h bva, bvb;
      {
        const _Float16* p0 = &Vt[cl * 40];
        const _Float16* p1 = &Vt[(16 + cl) * 40];
#pragma unroll
        for (int t = 0; t < 8; ++t) { bva[t] = p0[ksel8 + t]; bva[8 + t] = p0[16 + ksel8 + t]; bvb[t] = p1[ksel8 + t]; bvb[8 + t] = p1[16 + ksel8 + t]; }
      }
      oa[qt] = WMMA_F16(pf, bva, oa[qt]);
      ob[qt] = WMMA_F16(pf, bvb, ob[qt]);
    }
  }

  // normalize and store O as f16
#pragma unroll
  for (int qt = 0; qt < 2; ++qt) {
    int qrow0 = (w + qt * 8) * 16;
#pragma unroll
    for (int r = 0; r < 8; ++r) {
      float inv = 1.0f / lrow[qt][r];
      int rowg = qrow0 + r + hi8;
      _Float16* op = o + ((size_t)n * LSEQ + rowg) * CM + h * HD;
      op[cl]      = (_Float16)(oa[qt][r] * inv);
      op[16 + cl] = (_Float16)(ob[qt][r] * inv);
    }
  }
}

// ---------------------------------------------------------------------------
// Kernel: out projection + bias + residual (f32 out)
// ---------------------------------------------------------------------------
__global__ __launch_bounds__(256) void out_gemm_kernel(
    const _Float16* __restrict__ A,   // o  [MTOT][CM]
    const _Float16* __restrict__ Bt,  // WoutT [CM][CM]
    const float* __restrict__ bvec,   // [CM]
    const float* __restrict__ resid,  // m  [MTOT][CM]
    float* __restrict__ Omat) {       // [MTOT][CM]
  __shared__ __attribute__((aligned(16))) _Float16 Alds[128 * 40];
  __shared__ __attribute__((aligned(16))) _Float16 Blds[64 * 40];
  int tid = threadIdx.x, lane = tid & 31, w = tid >> 5;
  int wm = w >> 1, wn = w & 1;
  int m0 = blockIdx.y * 128, n0 = blockIdx.x * 64;

  v8f acc[2][2];
#pragma unroll
  for (int i = 0; i < 2; ++i)
#pragma unroll
    for (int j = 0; j < 2; ++j)
#pragma unroll
      for (int r = 0; r < 8; ++r) acc[i][j][r] = 0.f;

  int ar = tid >> 1, ah = (tid & 1) * 16;
  int br = tid >> 3, bc = (tid & 7) * 8;
  int rl = lane & 15, ksel8 = (lane >> 4) << 3, ksel16 = (lane >> 4) << 4;

  for (int k0 = 0; k0 < CM; k0 += 32) {
    __syncthreads();
    {
      const _Float16* s = A + (size_t)(m0 + ar) * CM + k0 + ah;
      uint4 t0 = *(const uint4*)s;
      uint4 t1 = *(const uint4*)(s + 8);
      *(uint4*)&Alds[ar * 40 + ah]     = t0;
      *(uint4*)&Alds[ar * 40 + ah + 8] = t1;
    }
    {
      union { uint4 u; _Float16 h[8]; } t;
      t.u = *(const uint4*)(Bt + (size_t)(k0 + br) * CM + n0 + bc);
#pragma unroll
      for (int u = 0; u < 8; ++u) Blds[(bc + u) * 40 + br] = t.h[u];
    }
    __syncthreads();

    v16h af[2], bf[2];
#pragma unroll
    for (int i = 0; i < 2; ++i) {
      const _Float16* p = &Alds[(wm * 32 + i * 16 + rl) * 40];
#pragma unroll
      for (int jj = 0; jj < 8; ++jj) { af[i][jj] = p[ksel8 + jj]; af[i][8 + jj] = p[16 + ksel8 + jj]; }
    }
#pragma unroll
    for (int j = 0; j < 2; ++j) {
      const _Float16* p = &Blds[(wn * 32 + j * 16 + rl) * 40];
#pragma unroll
      for (int t = 0; t < 8; ++t) { bf[j][t] = p[ksel8 + t]; bf[j][8 + t] = p[16 + ksel8 + t]; }
    }
#pragma unroll
    for (int i = 0; i < 2; ++i)
#pragma unroll
      for (int j = 0; j < 2; ++j)
        acc[i][j] = WMMA_F16(af[i], bf[j], acc[i][j]);
  }

  int hi8 = (lane >> 4) * 8;
#pragma unroll
  for (int i = 0; i < 2; ++i)
#pragma unroll
    for (int j = 0; j < 2; ++j) {
      int colg = n0 + wn * 32 + j * 16 + rl;
      float bv = bvec[colg];
#pragma unroll
      for (int rr = 0; rr < 8; ++rr) {
        int rowg = m0 + wm * 32 + i * 16 + rr + hi8;
        size_t off = (size_t)rowg * CM + colg;
        Omat[off] = resid[off] + acc[i][j][rr] + bv;
      }
    }
}

// ---------------------------------------------------------------------------
extern "C" void kernel_launch(void* const* d_in, const int* in_sizes, int n_in,
                              void* d_out, int out_size, void* d_ws, size_t ws_size,
                              hipStream_t stream) {
  (void)in_sizes; (void)n_in; (void)out_size; (void)ws_size;
  const float* m          = (const float*)d_in[0];
  const float* z          = (const float*)d_in[1];
  const float* ln_m_g     = (const float*)d_in[2];
  const float* ln_m_b     = (const float*)d_in[3];
  const float* ln_z_g     = (const float*)d_in[4];
  const float* ln_z_b     = (const float*)d_in[5];
  const float* w_pair     = (const float*)d_in[6];
  const float* in_proj_w  = (const float*)d_in[7];
  const float* in_proj_b  = (const float*)d_in[8];
  const float* out_proj_w = (const float*)d_in[9];
  const float* out_proj_b = (const float*)d_in[10];
  float* out = (float*)d_out;

  size_t off = 0;
  char* base = (char*)d_ws;
  auto alloc = [&](size_t bytes) -> char* {
    char* p = base + off;
    off += (bytes + 255) & ~(size_t)255;
    return p;
  };
  float*    bias  = (float*)alloc((size_t)NH * LSEQ * LSEQ * 4);     // 2 MB
  _Float16* mn_o  = (_Float16*)alloc((size_t)MTOT * CM * 2);         // 16 MB (mn, then reused for o)
  _Float16* wqkvT = (_Float16*)alloc((size_t)CM * EQKV * 2);         // 0.4 MB
  _Float16* woutT = (_Float16*)alloc((size_t)CM * CM * 2);           // 0.13 MB
  _Float16* qkvb  = (_Float16*)alloc((size_t)MTOT * EQKV * 2);       // 50 MB

  convert_wT_kernel<<<(EQKV * CM + 255) / 256, 256, 0, stream>>>(in_proj_w, wqkvT, EQKV, CM);
  convert_wT_kernel<<<(CM * CM + 255) / 256, 256, 0, stream>>>(out_proj_w, woutT, CM, CM);
  pair_bias_kernel<<<(LSEQ * LSEQ) / 128, 128, 0, stream>>>(z, ln_z_g, ln_z_b, w_pair, bias);
  ln_m_kernel<<<MTOT / 8, 256, 0, stream>>>(m, ln_m_g, ln_m_b, mn_o);
  qkv_gemm_kernel<<<dim3(EQKV / 64, MTOT / 128), 256, 0, stream>>>(mn_o, wqkvT, in_proj_b, qkvb);
  // mn no longer needed: reuse its buffer for attention output o
  attn_kernel<<<NROW * NH, 256, 0, stream>>>(qkvb, bias, mn_o);
  out_gemm_kernel<<<dim3(CM / 64, MTOT / 128), 256, 0, stream>>>(mn_o, woutT, out_proj_b, m, out);
}
